// AtlasTransformerV4_64355789963690
// MI455X (gfx1250) — matrix operand order repaired and checked by reading, not compiled, not run on hardware
//
#include <hip/hip_runtime.h>
#include <hip/hip_bf16.h>
#include <math.h>

// ---------------------------------------------------------------------------
// CDNA5 / gfx1250 implementation: all heavy GEMMs + attention use
// v_wmma_f32_16x16x32_bf16 (wave32 fragments per ISA 7.12.2).
// GEMM uses 32x64 per-wave register blocking (8 accumulators) so each
// B-fragment load feeds two WMMAs; global_prefetch_b8 covers the streams.
// ---------------------------------------------------------------------------

typedef __bf16 bf16;
typedef __attribute__((ext_vector_type(16))) __bf16 v16bf;
typedef __attribute__((ext_vector_type(8)))  float  v8f;

union FragU { v16bf v; uint4 q[2]; };

__device__ __forceinline__ v8f wmma_bf16(v16bf a, v16bf b, v8f c) {
  return __builtin_amdgcn_wmma_f32_16x16x32_bf16(false, a, false, b, (short)0, c,
                                                 false, false);
}

__device__ __forceinline__ float eluf(float x) {
  return x > 0.f ? x : (__expf(x) - 1.f);
}

__device__ __forceinline__ float block_sum(float v, float* sb) {
  int t = threadIdx.x;
  sb[t] = v; __syncthreads();
  for (int s = blockDim.x >> 1; s > 0; s >>= 1) {
    if (t < s) sb[t] += sb[t + s];
    __syncthreads();
  }
  float r = sb[0]; __syncthreads(); return r;
}

__device__ __forceinline__ float block_maxr(float v, float* sb) {
  int t = threadIdx.x;
  sb[t] = v; __syncthreads();
  for (int s = blockDim.x >> 1; s > 0; s >>= 1) {
    if (t < s) sb[t] = fmaxf(sb[t], sb[t + s]);
    __syncthreads();
  }
  float r = sb[0]; __syncthreads(); return r;
}

// ---------------------------------------------------------------------------
// Elementwise converters
// ---------------------------------------------------------------------------
__global__ void cvt_kernel(const float* __restrict__ s, bf16* __restrict__ d, int n) {
  int i = blockIdx.x * blockDim.x + threadIdx.x;
  if (i < n) d[i] = (bf16)s[i];
}

// Wt[n*K + k] = W[k*N + n]  (bf16 transposed weights -> contiguous B-fragments)
__global__ void trans_cvt_kernel(const float* __restrict__ W, bf16* __restrict__ Wt,
                                 int K, int N) {
  int i = blockIdx.x * blockDim.x + threadIdx.x;
  if (i < K * N) {
    int n = i / K, kk = i % K;
    Wt[i] = (bf16)W[(size_t)kk * N + n];
  }
}

// ---------------------------------------------------------------------------
// Generic WMMA GEMM: C(M x N) = A(M x K, bf16 row-major) * Wt(N x K, bf16)^T
// Block = 128 threads (4 waves), tile 128x64; each wave owns a 32x64 strip
// (two 16-row fragment groups) so every B fragment is reused twice.
// ---------------------------------------------------------------------------
enum { EPI_RAW = 0, EPI_Q, EPI_K, EPI_V, EPI_RESID, EPI_ELU };

template <int EPI>
__global__ __launch_bounds__(128) void gemm_bf16_kernel(
    const bf16* __restrict__ A, const bf16* __restrict__ Wt,
    const float* __restrict__ bias, const float* __restrict__ kmod,
    float* __restrict__ outf, bf16* __restrict__ outb, int M, int K, int N) {
  const v8f vzero = {0.f, 0.f, 0.f, 0.f, 0.f, 0.f, 0.f, 0.f};
  int wave = threadIdx.x >> 5;
  int lane = threadIdx.x & 31;
  int l16 = lane & 15, hi = lane >> 4;
  int row0 = blockIdx.x * 128 + wave * 32;
  int col0 = blockIdx.y * 64;
  v8f c[2][4];
#pragma unroll
  for (int g = 0; g < 2; ++g)
#pragma unroll
    for (int j = 0; j < 4; ++j) c[g][j] = vzero;

  const bf16* ap0 = A + (size_t)(row0 + l16) * K + hi * 8;
  const bf16* ap1 = ap0 + (size_t)16 * K;
  for (int kb = 0; kb < K; kb += 32) {
    FragU fa0, fa1;
    fa0.q[0] = *(const uint4*)(ap0 + kb);
    fa0.q[1] = *(const uint4*)(ap0 + kb + 16);
    fa1.q[0] = *(const uint4*)(ap1 + kb);
    fa1.q[1] = *(const uint4*)(ap1 + kb + 16);
    // speculative prefetch of the next A k-chunk (streamed exactly once)
    __builtin_prefetch(ap0 + kb + 32, 0, 0);
    __builtin_prefetch(ap1 + kb + 32, 0, 0);
#pragma unroll
    for (int j = 0; j < 4; ++j) {
      FragU fb;
      const bf16* bp = Wt + (size_t)(col0 + j * 16 + l16) * K + kb + hi * 8;
      fb.q[0] = *(const uint4*)bp;
      fb.q[1] = *(const uint4*)(bp + 16);
      c[0][j] = wmma_bf16(fa0.v, fb.v, c[0][j]);
      c[1][j] = wmma_bf16(fa1.v, fb.v, c[1][j]);
    }
  }
#pragma unroll
  for (int g = 0; g < 2; ++g) {
#pragma unroll
    for (int j = 0; j < 4; ++j) {
      int col = col0 + j * 16 + l16;
      float bcol = bias[col];
#pragma unroll
      for (int r = 0; r < 8; ++r) {
        int row = row0 + g * 16 + r + hi * 8;
        float v = c[g][j][r] + bcol;
        if (EPI == EPI_RAW) {
          outf[(size_t)row * N + col] = v;
        } else if (EPI == EPI_RESID) {
          outf[(size_t)row * N + col] += v;
        } else if (EPI == EPI_ELU) {
          outb[(size_t)row * N + col] = (bf16)eluf(v);
        } else {
          int b = row >> 10, n = row & 1023;
          int h = col >> 6, d = col & 63;
          if (EPI == EPI_K) v += kmod[b * 256 + col];
          if (EPI == EPI_V)
            outb[((size_t)((b * 4 + h) * 64 + d)) * 1024 + n] = (bf16)v;  // V^T
          else
            outb[((size_t)((b * 4 + h) * 1024 + n)) * 64 + d] = (bf16)v;  // Q / K
        }
      }
    }
  }
}

// ---------------------------------------------------------------------------
// Flash attention, D=64, N=1024. Grid(x=16 query tiles, y=B*H=64), 128 thr.
// q,k: (BH, N, 64) bf16 ; vT: (BH, 64, N) bf16 ; out: (B, N, 256) bf16.
// ---------------------------------------------------------------------------
__global__ __launch_bounds__(128) void attn_kernel(
    const bf16* __restrict__ q, const bf16* __restrict__ k,
    const bf16* __restrict__ vT, const int* __restrict__ mask,
    const float* __restrict__ hbias, bf16* __restrict__ outb, float scale) {
  __shared__ __attribute__((aligned(16))) bf16 Pst[4][16][32];
  const v8f vzero = {0.f, 0.f, 0.f, 0.f, 0.f, 0.f, 0.f, 0.f};
  int bh = blockIdx.y;
  int b = bh >> 2, h = bh & 3;
  int wave = threadIdx.x >> 5, lane = threadIdx.x & 31;
  int l16 = lane & 15, hi = lane >> 4;
  int qrow = blockIdx.x * 64 + wave * 16;

  // preload Q fragments (16 rows x K=64 -> two k-steps of 32)
  const bf16* qbase = q + ((size_t)bh * 1024 + qrow + l16) * 64 + hi * 8;
  FragU aq0, aq1;
  aq0.q[0] = *(const uint4*)qbase;        aq0.q[1] = *(const uint4*)(qbase + 16);
  aq1.q[0] = *(const uint4*)(qbase + 32); aq1.q[1] = *(const uint4*)(qbase + 48);

  float hb = hbias[bh];
  float m[8], l[8];
  v8f o[4]; o[0] = o[1] = o[2] = o[3] = vzero;
#pragma unroll
  for (int r = 0; r < 8; ++r) { m[r] = -1e30f; l[r] = 0.f; }
  const int* mrow = mask + b * 1024;

  for (int kt = 0; kt < 1024; kt += 32) {
    v8f s0 = vzero, s1 = vzero;
    {
      const bf16* kp0 = k + ((size_t)bh * 1024 + kt + l16) * 64 + hi * 8;
      // prefetch next key tile (32 rows x 64 cols ahead)
      __builtin_prefetch(kp0 + 32 * 64, 0, 0);
      FragU f0, f1;
      f0.q[0] = *(const uint4*)kp0;        f0.q[1] = *(const uint4*)(kp0 + 16);
      f1.q[0] = *(const uint4*)(kp0 + 32); f1.q[1] = *(const uint4*)(kp0 + 48);
      s0 = wmma_bf16(aq0.v, f0.v, s0);
      s0 = wmma_bf16(aq1.v, f1.v, s0);
      const bf16* kp1 = kp0 + 16 * 64;
      f0.q[0] = *(const uint4*)kp1;        f0.q[1] = *(const uint4*)(kp1 + 16);
      f1.q[0] = *(const uint4*)(kp1 + 32); f1.q[1] = *(const uint4*)(kp1 + 48);
      s1 = wmma_bf16(aq0.v, f0.v, s1);
      s1 = wmma_bf16(aq1.v, f1.v, s1);
    }
    int mk0 = mrow[kt + l16];
    int mk1 = mrow[kt + 16 + l16];
#pragma unroll
    for (int r = 0; r < 8; ++r) {
      float e0 = mk0 ? (s0[r] * scale + hb) : -1e30f;
      float e1 = mk1 ? (s1[r] * scale + hb) : -1e30f;
      float tm = fmaxf(e0, e1);
#pragma unroll
      for (int d = 1; d < 16; d <<= 1) tm = fmaxf(tm, __shfl_xor(tm, d, 32));
      float mn = fmaxf(m[r], tm);
      float alpha = __expf(m[r] - mn);
      float p0 = __expf(e0 - mn), p1 = __expf(e1 - mn);
      float ps = p0 + p1;
#pragma unroll
      for (int d = 1; d < 16; d <<= 1) ps += __shfl_xor(ps, d, 32);
      l[r] = l[r] * alpha + ps;
      m[r] = mn;
#pragma unroll
      for (int t = 0; t < 4; ++t) o[t][r] = o[t][r] * alpha;
      int prow = r + hi * 8;  // C-frag row owned by this lane half
      Pst[wave][prow][l16] = (bf16)p0;
      Pst[wave][prow][16 + l16] = (bf16)p1;
    }
    // relayout P: C-fragment -> A-fragment through per-wave LDS (in-order DS)
    FragU pa;
    const bf16* pp = &Pst[wave][l16][hi * 8];
    pa.q[0] = *(const uint4*)pp;
    pa.q[1] = *(const uint4*)(pp + 16);
#pragma unroll
    for (int t = 0; t < 4; ++t) {
      FragU fv;
      const bf16* vp = vT + ((size_t)bh * 64 + t * 16 + l16) * 1024 + kt + hi * 8;
      __builtin_prefetch(vp + 32, 0, 0);  // next V tile along N
      fv.q[0] = *(const uint4*)vp;
      fv.q[1] = *(const uint4*)(vp + 16);
      o[t] = wmma_bf16(pa.v, fv.v, o[t]);
    }
  }
#pragma unroll
  for (int t = 0; t < 4; ++t) {
#pragma unroll
    for (int r = 0; r < 8; ++r) {
      int n = qrow + r + hi * 8;
      float val = o[t][r] / l[r];
      outb[((size_t)b * 1024 + n) * 256 + h * 64 + t * 16 + l16] = (bf16)val;
    }
  }
}

// ---------------------------------------------------------------------------
// Small conditioning / encoder / readout kernels (B=16 rows -> VALU is fine)
// ---------------------------------------------------------------------------
__global__ void fill_treat_kernel(const float* tv, const float* tm, const float* tu,
                                  float* filled) {
  int i = blockIdx.x * blockDim.x + threadIdx.x;
  if (i < 16 * 11) {
    int j = i % 11;
    float mk = tm[i];
    filled[i] = tv[i] * mk + tu[j] * (1.f - mk);
  }
}

__global__ void scalar_feats_kernel(const float* clin, const float* atl,
                                    const float* ageW, const float* ageB,
                                    const float* sexW, const float* sexB,
                                    const float* atW, const float* atB,
                                    float* age_mod, float* sexf, float* atlf) {
  int b = blockIdx.x, j = threadIdx.x;
  age_mod[b * 256 + j] = eluf(clin[b * 2] * ageW[j] + ageB[j]);
  if (j < 64) {
    sexf[b * 64 + j] = clin[b * 2 + 1] * sexW[j] + sexB[j];
    atlf[b * 64 + j] = eluf(atl[b] * atW[j] + atB[j]);
  }
}

// lin -> LN -> elu -> lin -> elu   (graph & treatment encoders)
__global__ void mlp2_kernel(const float* __restrict__ in, int inDim,
                            const float* W1, const float* b1, const float* lg,
                            const float* lb, const float* W2, const float* b2,
                            float* out) {
  __shared__ float sb[256];
  __shared__ float hbuf[256];
  int b = blockIdx.x, j = threadIdx.x;
  float acc = b1[j];
  for (int kk = 0; kk < inDim; ++kk) acc += in[b * inDim + kk] * W1[kk * 256 + j];
  float mean = block_sum(acc, sb) * (1.f / 256.f);
  float dv = acc - mean;
  float var = block_sum(dv * dv, sb) * (1.f / 256.f);
  hbuf[j] = eluf(dv * rsqrtf(var + 1e-5f) * lg[j] + lb[j]);
  __syncthreads();
  float a2 = b2[j];
  for (int kk = 0; kk < 256; ++kk) a2 += hbuf[kk] * W2[kk * 256 + j];
  out[b * 256 + j] = eluf(a2);
}

// per-layer key-mod vector (B,256) and per-head attention bias (B,4)
__global__ void cond_kernel(const float* g, const float* t, const int* ct,
                            const float* ckm, const float* cab,
                            const float* gkmW, const float* gkmB,
                            const float* tkmW, const float* tkmB,
                            const float* gabW1, const float* gabB1,
                            const float* gabW2, const float* gabB2,
                            const float* tabW1, const float* tabB1,
                            const float* tabW2, const float* tabB2,
                            float* kmod, float* hbias) {
  int b = blockIdx.x, j = threadIdx.x, c = ct[b];
  float a1 = gkmB[j], a2 = tkmB[j];
  for (int kk = 0; kk < 256; ++kk) {
    a1 += g[b * 256 + kk] * gkmW[kk * 256 + j];
    a2 += t[b * 256 + kk] * tkmW[kk * 256 + j];
  }
  kmod[b * 256 + j] = ckm[c * 256 + j] + eluf(a1) + eluf(a2);
  if (j < 4) {
    float acc = cab[c * 4 + j] + gabB2[j] + tabB2[j];
    for (int i = 0; i < 8; ++i) {
      float hg = gabB1[i], ht = tabB1[i];
      for (int kk = 0; kk < 256; ++kk) {
        hg += g[b * 256 + kk] * gabW1[kk * 8 + i];
        ht += t[b * 256 + kk] * tabW1[kk * 8 + i];
      }
      acc += eluf(hg) * gabW2[i * 4 + j] + eluf(ht) * tabW2[i * 4 + j];
    }
    hbias[b * 4 + j] = acc;
  }
}

// LN(x) -> bf16 (pre-attention / pre-FFN)
__global__ void ln_bf16_kernel(const float* __restrict__ x, const float* g,
                               const float* bb, bf16* __restrict__ out) {
  __shared__ float sb[256];
  int row = blockIdx.x, j = threadIdx.x;
  float v = x[(size_t)row * 256 + j];
  float mean = block_sum(v, sb) * (1.f / 256.f);
  float dv = v - mean;
  float var = block_sum(dv * dv, sb) * (1.f / 256.f);
  out[(size_t)row * 256 + j] = (bf16)(dv * rsqrtf(var + 1e-5f) * g[j] + bb[j]);
}

// node encoder epilogue: x = elu(LN(raw)) * (1 + age_mod) + pos_emb
__global__ void node_post_kernel(const float* __restrict__ tmpf, const float* g,
                                 const float* bb, const float* age_mod,
                                 const float* pos_emb, const int* pos_ids,
                                 float* __restrict__ x) {
  __shared__ float sb[256];
  int row = blockIdx.x, j = threadIdx.x;
  int b = row >> 10, n = row & 1023;
  float v = tmpf[(size_t)row * 256 + j];
  float mean = block_sum(v, sb) * (1.f / 256.f);
  float dv = v - mean;
  float var = block_sum(dv * dv, sb) * (1.f / 256.f);
  float y = eluf(dv * rsqrtf(var + 1e-5f) * g[j] + bb[j]);
  y = y * (1.f + age_mod[b * 256 + j]) + pos_emb[pos_ids[b * 1024 + n] * 256 + j];
  x[(size_t)row * 256 + j] = y;
}

__global__ void pool_logits_kernel(const float* __restrict__ x, const float* pW1,
                                   const float* pb1, const float* pW2,
                                   const float* pb2, const int* mask,
                                   float* wlog) {
  __shared__ float sb[128];
  int row = blockIdx.x, i = threadIdx.x;
  float acc = pb1[i];
  for (int kk = 0; kk < 256; ++kk) acc += x[(size_t)row * 256 + kk] * pW1[kk * 128 + i];
  float hv = tanhf(acc) * pW2[i];
  float s = block_sum(hv, sb);
  if (i == 0) wlog[row] = mask[row] ? (s + pb2[0]) : -1e30f;
}

__global__ void pool_softmax_kernel(const float* __restrict__ x,
                                    const float* __restrict__ wlog, float* pemb) {
  __shared__ float sb[256];
  int b = blockIdx.x, j = threadIdx.x;
  float mx = -1e30f;
  for (int n = j; n < 1024; n += 256) mx = fmaxf(mx, wlog[b * 1024 + n]);
  mx = block_maxr(mx, sb);
  float se = 0.f;
  for (int n = j; n < 1024; n += 256) se += __expf(wlog[b * 1024 + n] - mx);
  se = block_sum(se, sb);
  float acc = 0.f;
  for (int n = 0; n < 1024; ++n) {
    float p = __expf(wlog[b * 1024 + n] - mx);
    acc += p * x[((size_t)b * 1024 + n) * 256 + j];
  }
  pemb[b * 256 + j] = acc / se;
}

__global__ void readout_kernel(const float* pemb, const float* sexf,
                               const float* atlf, const float* g, const float* t,
                               const float* roW1, const float* rob1,
                               const float* roW2, const float* rob2, float* out) {
  __shared__ float sb[256];
  __shared__ float cat[896];
  int b = blockIdx.x, j = threadIdx.x;
  cat[j] = pemb[b * 256 + j];
  if (j < 64) {
    cat[256 + j] = sexf[b * 64 + j];
    cat[320 + j] = atlf[b * 64 + j];
  }
  cat[384 + j] = g[b * 256 + j];
  cat[640 + j] = t[b * 256 + j];
  __syncthreads();
  float a = rob1[j];
  for (int c = 0; c < 896; ++c) a += cat[c] * roW1[c * 256 + j];
  float hv = eluf(a) * roW2[j];
  float s = block_sum(hv, sb);
  if (j == 0) out[b] = s + rob2[0];
}

// ---------------------------------------------------------------------------
// Host orchestration
// ---------------------------------------------------------------------------
enum {
  IN_NODE_FEAT = 0, IN_MASK, IN_POS, IN_CANCER, IN_CLIN, IN_ATLAS, IN_GF,
  IN_TVEC, IN_TMASK,
  P_AGE_W, P_AGE_B, P_NODE_W, P_NODE_B, P_NODE_LN_G, P_NODE_LN_B, P_POS_EMB,
  P_GE_W1, P_GE_B1, P_GE_LN_G, P_GE_LN_B, P_GE_W2, P_GE_B2,
  P_TU, P_TE_W1, P_TE_B1, P_TE_LN_G, P_TE_LN_B, P_TE_W2, P_TE_B2,
  P_L0  // = 29
};
static const int LSTRIDE = 30;
enum {
  L_QW = 0, L_QB, L_KW, L_KB, L_VW, L_VB, L_OW, L_OB, L_CAB, L_CKM,
  L_GAB_W1, L_GAB_B1, L_GAB_W2, L_GAB_B2, L_GKM_W, L_GKM_B, L_TKM_W, L_TKM_B,
  L_TAB_W1, L_TAB_B1, L_TAB_W2, L_TAB_B2, L_N1G, L_N1B, L_N2G, L_N2B,
  L_F1W, L_F1B, L_F2W, L_F2B
};

extern "C" void kernel_launch(void* const* d_in, const int* in_sizes, int n_in,
                              void* d_out, int out_size, void* d_ws,
                              size_t ws_size, hipStream_t stream) {
  (void)in_sizes; (void)n_in; (void)out_size; (void)ws_size;
  auto F = [&](int i) { return (const float*)d_in[i]; };
  auto Iv = [&](int i) { return (const int*)d_in[i]; };
  auto LP = [&](int l, int f) { return (const float*)d_in[P_L0 + l * LSTRIDE + f]; };

  const int M = 16 * 1024;  // tokens

  char* ws = (char*)d_ws;
  size_t off = 0;
  auto alloc = [&](size_t bytes) -> char* {
    char* p = ws + off;
    off += (bytes + 255) & ~(size_t)255;
    return p;
  };

  float* x   = (float*)alloc((size_t)M * 256 * 4);   // residual stream
  bf16* xnb  = (bf16*)alloc((size_t)M * 256 * 2);    // LN(x) bf16
  bf16* qb   = (bf16*)alloc((size_t)M * 256 * 2);
  bf16* kb   = (bf16*)alloc((size_t)M * 256 * 2);
  bf16* vtb  = (bf16*)alloc((size_t)M * 256 * 2);
  char* bufc = alloc((size_t)M * 256 * 4);           // tmpf / attno / ffn hidden
  float* tmpf = (float*)bufc;
  bf16* attno = (bf16*)bufc;
  bf16* hhid  = (bf16*)bufc;
  bf16* nfb  = (bf16*)alloc((size_t)M * 64 * 2);

  bf16* WnT = (bf16*)alloc(64 * 256 * 2);
  bf16 *WqT[2], *WkT[2], *WvT[2], *WoT[2], *Wf1T[2], *Wf2T[2];
  for (int l = 0; l < 2; ++l) {
    WqT[l]  = (bf16*)alloc(256 * 256 * 2);
    WkT[l]  = (bf16*)alloc(256 * 256 * 2);
    WvT[l]  = (bf16*)alloc(256 * 256 * 2);
    WoT[l]  = (bf16*)alloc(256 * 256 * 2);
    Wf1T[l] = (bf16*)alloc(512 * 256 * 2);
    Wf2T[l] = (bf16*)alloc(256 * 512 * 2);
  }
  float* filled  = (float*)alloc(16 * 11 * 4);
  float* age_mod = (float*)alloc(16 * 256 * 4);
  float* sexf    = (float*)alloc(16 * 64 * 4);
  float* atlf    = (float*)alloc(16 * 64 * 4);
  float* gvec    = (float*)alloc(16 * 256 * 4);
  float* tvecE   = (float*)alloc(16 * 256 * 4);
  float* kmodL[2]; float* hbiasL[2];
  for (int l = 0; l < 2; ++l) {
    kmodL[l]  = (float*)alloc(16 * 256 * 4);
    hbiasL[l] = (float*)alloc(16 * 4 * 4);
  }
  float* wlog = (float*)alloc(16 * 1024 * 4);
  float* pemb = (float*)alloc(16 * 256 * 4);

  // ---- prep: bf16 conversions / weight transposes ----
  int nfTot = M * 64;
  cvt_kernel<<<(nfTot + 255) / 256, 256, 0, stream>>>(F(IN_NODE_FEAT), nfb, nfTot);
  trans_cvt_kernel<<<(64 * 256 + 255) / 256, 256, 0, stream>>>(F(P_NODE_W), WnT, 64, 256);
  for (int l = 0; l < 2; ++l) {
    trans_cvt_kernel<<<256, 256, 0, stream>>>(LP(l, L_QW), WqT[l], 256, 256);
    trans_cvt_kernel<<<256, 256, 0, stream>>>(LP(l, L_KW), WkT[l], 256, 256);
    trans_cvt_kernel<<<256, 256, 0, stream>>>(LP(l, L_VW), WvT[l], 256, 256);
    trans_cvt_kernel<<<256, 256, 0, stream>>>(LP(l, L_OW), WoT[l], 256, 256);
    trans_cvt_kernel<<<512, 256, 0, stream>>>(LP(l, L_F1W), Wf1T[l], 256, 512);
    trans_cvt_kernel<<<512, 256, 0, stream>>>(LP(l, L_F2W), Wf2T[l], 512, 256);
  }

  // ---- conditioning (B=16) ----
  fill_treat_kernel<<<1, 256, 0, stream>>>(F(IN_TVEC), F(IN_TMASK), F(P_TU), filled);
  scalar_feats_kernel<<<16, 256, 0, stream>>>(F(IN_CLIN), F(IN_ATLAS), F(P_AGE_W),
      F(P_AGE_B), F(93), F(94), F(95), F(96), age_mod, sexf, atlf);
  mlp2_kernel<<<16, 256, 0, stream>>>(F(IN_GF), 128, F(P_GE_W1), F(P_GE_B1),
      F(P_GE_LN_G), F(P_GE_LN_B), F(P_GE_W2), F(P_GE_B2), gvec);
  mlp2_kernel<<<16, 256, 0, stream>>>(filled, 11, F(P_TE_W1), F(P_TE_B1),
      F(P_TE_LN_G), F(P_TE_LN_B), F(P_TE_W2), F(P_TE_B2), tvecE);
  for (int l = 0; l < 2; ++l) {
    cond_kernel<<<16, 256, 0, stream>>>(gvec, tvecE, Iv(IN_CANCER),
        LP(l, L_CKM), LP(l, L_CAB), LP(l, L_GKM_W), LP(l, L_GKM_B),
        LP(l, L_TKM_W), LP(l, L_TKM_B), LP(l, L_GAB_W1), LP(l, L_GAB_B1),
        LP(l, L_GAB_W2), LP(l, L_GAB_B2), LP(l, L_TAB_W1), LP(l, L_TAB_B1),
        LP(l, L_TAB_W2), LP(l, L_TAB_B2), kmodL[l], hbiasL[l]);
  }

  // ---- node encoder: WMMA GEMM then LN/ELU/age/pos epilogue ----
  gemm_bf16_kernel<EPI_RAW><<<dim3(M / 128, 4), 128, 0, stream>>>(
      nfb, WnT, F(P_NODE_B), nullptr, tmpf, nullptr, M, 64, 256);
  node_post_kernel<<<M, 256, 0, stream>>>(tmpf, F(P_NODE_LN_G), F(P_NODE_LN_B),
      age_mod, F(P_POS_EMB), Iv(IN_POS), x);

  // ---- transformer layers ----
  for (int l = 0; l < 2; ++l) {
    ln_bf16_kernel<<<M, 256, 0, stream>>>(x, LP(l, L_N1G), LP(l, L_N1B), xnb);
    gemm_bf16_kernel<EPI_Q><<<dim3(M / 128, 4), 128, 0, stream>>>(
        xnb, WqT[l], LP(l, L_QB), nullptr, nullptr, qb, M, 256, 256);
    gemm_bf16_kernel<EPI_K><<<dim3(M / 128, 4), 128, 0, stream>>>(
        xnb, WkT[l], LP(l, L_KB), kmodL[l], nullptr, kb, M, 256, 256);
    gemm_bf16_kernel<EPI_V><<<dim3(M / 128, 4), 128, 0, stream>>>(
        xnb, WvT[l], LP(l, L_VB), nullptr, nullptr, vtb, M, 256, 256);
    attn_kernel<<<dim3(16, 64), 128, 0, stream>>>(qb, kb, vtb, Iv(IN_MASK),
        hbiasL[l], attno, 0.125f);
    gemm_bf16_kernel<EPI_RESID><<<dim3(M / 128, 4), 128, 0, stream>>>(
        attno, WoT[l], LP(l, L_OB), nullptr, x, nullptr, M, 256, 256);
    ln_bf16_kernel<<<M, 256, 0, stream>>>(x, LP(l, L_N2G), LP(l, L_N2B), xnb);
    gemm_bf16_kernel<EPI_ELU><<<dim3(M / 128, 8), 128, 0, stream>>>(
        xnb, Wf1T[l], LP(l, L_F1B), nullptr, nullptr, hhid, M, 256, 512);
    gemm_bf16_kernel<EPI_RESID><<<dim3(M / 128, 4), 128, 0, stream>>>(
        hhid, Wf2T[l], LP(l, L_F2B), nullptr, x, nullptr, M, 512, 256);
  }

  // ---- attention pooling + readout ----
  pool_logits_kernel<<<M, 128, 0, stream>>>(x, F(89), F(90), F(91), F(92),
                                            Iv(IN_MASK), wlog);
  pool_softmax_kernel<<<16, 256, 0, stream>>>(x, wlog, pemb);
  readout_kernel<<<16, 256, 0, stream>>>(pemb, sexf, atlf, gvec, tvecE,
                                         F(97), F(98), F(99), F(100),
                                         (float*)d_out);
}
